// GAT_29953101922730
// MI455X (gfx1250) — matrix operand order repaired and checked by reading, not compiled, not run on hardware
//
#include <hip/hip_runtime.h>

// ---------------------------------------------------------------------------
// CDNA5 (gfx1250) GAT: 2-layer GATConv + LN/ReLU + LN/log_softmax
// GEMMs run on v_wmma_f32_16x16x32_bf16 (wave32, 16x16 tiles, fp32 accum).
// Edge softmax/aggregation: wave-per-edge coalesced gather + f32 atomics.
// ---------------------------------------------------------------------------

typedef __attribute__((ext_vector_type(16))) __bf16 v16bf;
typedef __attribute__((ext_vector_type(8)))  float  v8f;

struct U128 { unsigned x, y, z, w; };

union FragAB { v16bf v; U128 q[2]; };
union FragC  { v8f   v; float f[8]; };

__device__ __forceinline__ unsigned short f2bf(float f) {
  unsigned u = __float_as_uint(f);
  return (unsigned short)((u + 0x7FFFu + ((u >> 16) & 1u)) >> 16);  // RNE
}

// order-preserving float -> uint map (for atomicMax-based segment max)
__device__ __forceinline__ unsigned fenc(float f) {
  unsigned u = __float_as_uint(f);
  return (u & 0x80000000u) ? ~u : (u ^ 0x80000000u);
}
__device__ __forceinline__ float fdec(unsigned e) {
  unsigned u = (e & 0x80000000u) ? (e ^ 0x80000000u) : ~e;
  return __uint_as_float(u);
}
#define ENC_NEG_INF 0x007FFFFFu   // fenc(-inf)

__device__ __forceinline__ float wredsum(float v) {
#pragma unroll
  for (int o = 16; o > 0; o >>= 1) v += __shfl_xor(v, o, 32);
  return v;
}
__device__ __forceinline__ float wredmax(float v) {
#pragma unroll
  for (int o = 16; o > 0; o >>= 1) v = fmaxf(v, __shfl_xor(v, o, 32));
  return v;
}

// ------------------------------- conversions -------------------------------

__global__ void k_f2bf(const float* __restrict__ in, unsigned short* __restrict__ out,
                       long long n) {
  long long t = (long long)blockIdx.x * blockDim.x + threadIdx.x;
  if (t < n) out[t] = f2bf(in[t]);
}

// Pack row-major W[K][Nw] (f32) into per-lane bf16 WMMA B-fragments.
// Tile (ntile,ktile) is a 32x16 (KxN) block; layout per ISA:
//   lane l: N = n0 + (l&15), khalf = l>>4,
//   half p (0..1) elem e holds K = p*16 + khalf*8 + e.
// Storage: [(ntile*KT+ktile)][half(0..1)][lane(0..31)][8 bf16]  (half0 = v0..3)
__global__ void k_packW(const float* __restrict__ W, unsigned short* __restrict__ out,
                        int K, int Nw) {
  int t = blockIdx.x * blockDim.x + threadIdx.x;
  int total = K * Nw;
  if (t >= total) return;
  int KT    = K >> 5;
  int tile  = t >> 9;        // 512 elems per tile
  int r     = t & 511;
  int half  = r >> 8;
  int lr    = r & 255;
  int lane  = lr >> 3;
  int e     = lr & 7;
  int ktile = tile % KT;
  int ntile = tile / KT;
  int kk = (ktile << 5) + ((lane >> 4) << 3) + (half << 4) + e;
  int n  = (ntile << 4) + (lane & 15);
  out[t] = f2bf(W[kk * Nw + n]);
}

__global__ void k_fill_u32(unsigned* __restrict__ p, unsigned val, long long n) {
  long long t = (long long)blockIdx.x * blockDim.x + threadIdx.x;
  if (t < n) p[t] = val;
}

// ---------------------------------- GEMM -----------------------------------
// C[M][Nw] (f32) = A[M][K] (bf16 row-major) * Bpacked.  One wave -> 16x64 tile.

__global__ void k_gemm_bf16(const unsigned short* __restrict__ A,
                            const unsigned short* __restrict__ Bp,
                            float* __restrict__ C,
                            long long M, int K, int Nw, long long totalWaves) {
  long long wid = ((long long)blockIdx.x * blockDim.x + threadIdx.x) >> 5;
  int lane = threadIdx.x & 31;
  if (wid >= totalWaves) return;

  int nquads = Nw >> 6;                       // 64-col quads per row of tiles
  long long mtile = wid / nquads;
  int nquad = (int)(wid % nquads);
  long long m0 = mtile << 4;
  int n0 = nquad << 6;
  int KT = K >> 5;

  long long mrow = m0 + (lane & 15);
  if (mrow >= M) mrow = M - 1;                // clamp (M is tile-aligned here)
  int kb = (lane >> 4) << 3;                  // ISA 16x32 bf16 A layout
  const unsigned short* arow = A + mrow * (long long)K + kb;

  FragC acc[4];
  v8f zero = {0.f, 0.f, 0.f, 0.f, 0.f, 0.f, 0.f, 0.f};
#pragma unroll
  for (int j = 0; j < 4; ++j) acc[j].v = zero;

  for (int kt = 0; kt < KT; ++kt) {
    FragAB a;
    a.q[0] = *(const U128*)(arow);            // K kb..kb+7
    a.q[1] = *(const U128*)(arow + 16);       // K kb+16..kb+23
    // WGP-scope prefetch two iterations (128B) ahead: consumed two
    // trips later, so fill all cache levels (locality 3 -> SCOPE_WGP), not GL2-only.
    __builtin_prefetch(arow + 64, 0, 3);
    arow += 32;
#pragma unroll
    for (int j = 0; j < 4; ++j) {
      long long ntile = (long long)(n0 >> 4) + j;
      const unsigned short* bp = Bp + ((ntile * KT + kt) * 2) * 256 + lane * 8;
      FragAB b;
      b.q[0] = *(const U128*)(bp);            // half 0 (VGPR 0..3)
      b.q[1] = *(const U128*)(bp + 256);      // half 1 (VGPR 4..7)
      acc[j].v = __builtin_amdgcn_wmma_f32_16x16x32_bf16(
          false, a.v, false, b.v, (short)0, acc[j].v, false, false);
    }
  }

  long long crow0 = m0 + ((lane >> 4) << 3);  // C layout: VGPR r -> M = r or 8+r
  int cn = lane & 15;
#pragma unroll
  for (int j = 0; j < 4; ++j) {
#pragma unroll
    for (int r = 0; r < 8; ++r) {
      long long row = crow0 + r;
      if (row < M) C[row * Nw + n0 + (j << 4) + cn] = acc[j].f[r];
    }
  }
}

// ------------------------------ attention prep -----------------------------
// es[n,h] = sum_c xh[n,h,c]*a_src[h,c];  ed likewise.  H = 1<<hshift.

__global__ void k_esed(const float* __restrict__ xh,
                       const float* __restrict__ as, const float* __restrict__ ad,
                       float* __restrict__ es, float* __restrict__ ed,
                       long long NH, int hshift, int C) {
  long long t = (long long)blockIdx.x * blockDim.x + threadIdx.x;
  if (t >= NH) return;
  long long n = t >> hshift;
  int h = (int)(t & ((1 << hshift) - 1));
  int HC = C << hshift;
  const float* p  = xh + n * HC + (long long)h * C;
  const float* pa = as + h * C;
  const float* pd = ad + h * C;
  float s = 0.f, d = 0.f;
  for (int c = 0; c < C; ++c) { float v = p[c]; s += v * pa[c]; d += v * pd[c]; }
  es[t] = s;
  ed[t] = d;
}

// ------------------------------ edge passes --------------------------------
// edge e < E: (src,dst) from edge_index; e >= E: self loop (e-E, e-E).

__global__ void k_edge_max(const long long* __restrict__ ei, long long E,
                           long long total, const float* __restrict__ es,
                           const float* __restrict__ ed, unsigned* __restrict__ m,
                           int hshift) {
  long long t = (long long)blockIdx.x * blockDim.x + threadIdx.x;
  if (t >= total) return;
  long long e = t >> hshift;
  int h = (int)(t & ((1 << hshift) - 1));
  int H = 1 << hshift;
  long long src, dst;
  if (e < E) { src = ei[e]; dst = ei[E + e]; } else { src = dst = e - E; }
  float s = es[src * H + h] + ed[dst * H + h];
  float lr = s > 0.f ? s : 0.2f * s;          // leaky_relu(0.2)
  atomicMax(&m[dst * H + h], fenc(lr));
}

__global__ void k_edge_sum(const long long* __restrict__ ei, long long E,
                           long long total, const float* __restrict__ es,
                           const float* __restrict__ ed,
                           const unsigned* __restrict__ m,
                           float* __restrict__ den, int hshift) {
  long long t = (long long)blockIdx.x * blockDim.x + threadIdx.x;
  if (t >= total) return;
  long long e = t >> hshift;
  int h = (int)(t & ((1 << hshift) - 1));
  int H = 1 << hshift;
  long long src, dst;
  if (e < E) { src = ei[e]; dst = ei[E + e]; } else { src = dst = e - E; }
  float s = es[src * H + h] + ed[dst * H + h];
  float lr = s > 0.f ? s : 0.2f * s;
  float w = __expf(lr - fdec(m[dst * H + h]));
  atomicAdd(&den[dst * H + h], w);
}

// one wave per edge; lane = channel within a 32-wide chunk; head uniform/iter
__global__ void k_edge_aggr(const long long* __restrict__ ei, long long E,
                            long long ET, const float* __restrict__ es,
                            const float* __restrict__ ed,
                            const unsigned* __restrict__ m,
                            const float* __restrict__ den,
                            const float* __restrict__ xh,
                            float* __restrict__ agg,
                            int H, int cshift, int iters) {
  long long wid = ((long long)blockIdx.x * blockDim.x + threadIdx.x) >> 5;
  int lane = threadIdx.x & 31;
  if (wid >= ET) return;
  long long src, dst;
  if (wid < E) { src = ei[wid]; dst = ei[E + wid]; } else { src = dst = wid - E; }
  int HC = iters << 5;
  for (int i = 0; i < iters; ++i) {
    int idx = (i << 5) + lane;
    int h = idx >> cshift;                    // uniform across the wave
    float s = es[src * H + h] + ed[dst * H + h];
    float lr = s > 0.f ? s : 0.2f * s;
    float w = __expf(lr - fdec(m[dst * H + h]));
    float alpha = w / den[dst * H + h];
    float v = xh[src * (long long)HC + idx];
    atomicAdd(&agg[dst * (long long)HC + idx], alpha * v);
  }
}

// ------------------------------ epilogues ----------------------------------
// wave per node, D = 256: +bias, LayerNorm, ReLU, emit bf16 for next GEMM.
__global__ void k_ln_relu_bf16(const float* __restrict__ agg,
                               const float* __restrict__ bias,
                               const float* __restrict__ g,
                               const float* __restrict__ be,
                               unsigned short* __restrict__ out, long long Nn) {
  long long wid = ((long long)blockIdx.x * blockDim.x + threadIdx.x) >> 5;
  int lane = threadIdx.x & 31;
  if (wid >= Nn) return;
  const float* p = agg + wid * 256 + lane * 8;
  int c0 = lane * 8;
  float v[8];
#pragma unroll
  for (int k = 0; k < 8; ++k) v[k] = p[k] + bias[c0 + k];
  float s = 0.f;
#pragma unroll
  for (int k = 0; k < 8; ++k) s += v[k];
  float mu = wredsum(s) * (1.0f / 256.0f);
  float q = 0.f;
#pragma unroll
  for (int k = 0; k < 8; ++k) { float d = v[k] - mu; q += d * d; }
  float inv = rsqrtf(wredsum(q) * (1.0f / 256.0f) + 1e-5f);
  unsigned pk[4];
#pragma unroll
  for (int k = 0; k < 4; ++k) {
    float y0 = fmaxf((v[2 * k]     - mu) * inv * g[c0 + 2 * k]     + be[c0 + 2 * k],     0.f);
    float y1 = fmaxf((v[2 * k + 1] - mu) * inv * g[c0 + 2 * k + 1] + be[c0 + 2 * k + 1], 0.f);
    pk[k] = (unsigned)f2bf(y0) | ((unsigned)f2bf(y1) << 16);
  }
  U128 r = {pk[0], pk[1], pk[2], pk[3]};
  *(U128*)(out + wid * 256 + lane * 8) = r;
}

// wave per node, D = 64: +bias, LayerNorm, log_softmax -> final output (f32)
__global__ void k_ln_logsm(const float* __restrict__ agg,
                           const float* __restrict__ bias,
                           const float* __restrict__ g,
                           const float* __restrict__ be,
                           float* __restrict__ out, long long Nn) {
  long long wid = ((long long)blockIdx.x * blockDim.x + threadIdx.x) >> 5;
  int lane = threadIdx.x & 31;
  if (wid >= Nn) return;
  int c = lane * 2;
  float x0 = agg[wid * 64 + c]     + bias[c];
  float x1 = agg[wid * 64 + c + 1] + bias[c + 1];
  float mu = wredsum(x0 + x1) * (1.0f / 64.0f);
  float d0 = x0 - mu, d1 = x1 - mu;
  float inv = rsqrtf(wredsum(d0 * d0 + d1 * d1) * (1.0f / 64.0f) + 1e-5f);
  float y0 = d0 * inv * g[c]     + be[c];
  float y1 = d1 * inv * g[c + 1] + be[c + 1];
  float mx = wredmax(fmaxf(y0, y1));
  float se = wredsum(__expf(y0 - mx) + __expf(y1 - mx));
  float ls = __logf(se) + mx;
  out[wid * 64 + c]     = y0 - ls;
  out[wid * 64 + c + 1] = y1 - ls;
}

// --------------------------------- driver ----------------------------------

extern "C" void kernel_launch(void* const* d_in, const int* in_sizes, int n_in,
                              void* d_out, int out_size, void* d_ws, size_t ws_size,
                              hipStream_t stream) {
  const float*     x   = (const float*)d_in[0];
  const long long* ei  = (const long long*)d_in[1];
  const float*     W1  = (const float*)d_in[2];
  const float*     as1 = (const float*)d_in[3];
  const float*     ad1 = (const float*)d_in[4];
  const float*     b1  = (const float*)d_in[5];
  const float*     g1  = (const float*)d_in[6];
  const float*     be1 = (const float*)d_in[7];
  const float*     W2  = (const float*)d_in[8];
  const float*     as2 = (const float*)d_in[9];
  const float*     ad2 = (const float*)d_in[10];
  const float*     b2  = (const float*)d_in[11];
  const float*     g2  = (const float*)d_in[12];
  const float*     be2 = (const float*)d_in[13];

  const int DIN = 256, HID = 256, OUTD = 64;
  const long long N  = (long long)in_sizes[0] / DIN;
  const long long E  = (long long)in_sizes[1] / 2;
  const long long ET = E + N;                // + self-loops

  // ---- workspace carve-out (bump allocator, 256B aligned) ----
  char* ws = (char*)d_ws;
  size_t off = 0;
  auto carve = [&](size_t bytes) -> char* {
    size_t o = (off + 255) & ~(size_t)255;
    off = o + bytes;
    return ws + o;
  };
  unsigned short* xbf  = (unsigned short*)carve((size_t)N * DIN * 2); // 51.2MB
  unsigned short* w1p  = (unsigned short*)carve((size_t)DIN * HID * 2);
  unsigned short* w2p  = (unsigned short*)carve((size_t)HID * OUTD * 2);
  float*    xh1  = (float*)carve((size_t)N * HID * 4);                // 102.4MB
  float*    es1  = (float*)carve((size_t)N * 8 * 4);
  float*    ed1  = (float*)carve((size_t)N * 8 * 4);
  unsigned* m1   = (unsigned*)carve((size_t)N * 8 * 4);
  float*    den1 = (float*)carve((size_t)N * 8 * 4);
  float*    agg1 = (float*)carve((size_t)N * HID * 4);                // 102.4MB
  float*    es2  = (float*)carve((size_t)N * 4);
  float*    ed2  = (float*)carve((size_t)N * 4);
  unsigned* m2   = (unsigned*)carve((size_t)N * 4);
  float*    den2 = (float*)carve((size_t)N * 4);
  // aliases: xbf region is dead after GEMM1 -> reuse for layer-2 activations
  float*    xh2  = (float*)xbf;                      // N*64*4 = 25.6MB
  float*    agg2 = (float*)xbf + (size_t)N * OUTD;   // next 25.6MB (fits in xbf)
  // xh1 region is dead after edge pass 3 -> reuse for bf16 hidden activations
  unsigned short* h1bf = (unsigned short*)xh1;

  const int B = 256;
  auto nblk = [](long long t, int b) { return (unsigned)((t + b - 1) / b); };

  // ---- init (capture-legal memset nodes + fill kernel) ----
  hipMemsetAsync(den1, 0, (size_t)N * 8 * 4, stream);
  hipMemsetAsync(agg1, 0, (size_t)N * HID * 4, stream);
  hipMemsetAsync(den2, 0, (size_t)N * 4, stream);
  k_fill_u32<<<nblk(N * 8, B), B, 0, stream>>>(m1, ENC_NEG_INF, N * 8);
  k_fill_u32<<<nblk(N, B), B, 0, stream>>>(m2, ENC_NEG_INF, N);

  // ---- convert inputs to bf16 / packed fragments ----
  k_f2bf<<<nblk(N * DIN, B), B, 0, stream>>>(x, xbf, N * DIN);
  k_packW<<<nblk(DIN * HID, B), B, 0, stream>>>(W1, w1p, DIN, HID);
  k_packW<<<nblk(HID * OUTD, B), B, 0, stream>>>(W2, w2p, HID, OUTD);

  // ---- layer 1: xh1 = x @ W1 (WMMA bf16) ----
  {
    long long mtiles = (N + 15) >> 4;
    long long waves = mtiles * (HID >> 6);
    k_gemm_bf16<<<nblk(waves * 32, B), B, 0, stream>>>(xbf, w1p, xh1, N, DIN, HID, waves);
  }
  k_esed<<<nblk(N * 8, B), B, 0, stream>>>(xh1, as1, ad1, es1, ed1, N * 8, 3, 32);
  k_edge_max<<<nblk(ET * 8, B), B, 0, stream>>>(ei, E, ET * 8, es1, ed1, m1, 3);
  k_edge_sum<<<nblk(ET * 8, B), B, 0, stream>>>(ei, E, ET * 8, es1, ed1, m1, den1, 3);
  k_edge_aggr<<<nblk(ET * 32, B), B, 0, stream>>>(ei, E, ET, es1, ed1, m1, den1,
                                                  xh1, agg1, 8, 5, 8);
  k_ln_relu_bf16<<<nblk(N * 32, B), B, 0, stream>>>(agg1, b1, g1, be1, h1bf, N);

  // ---- layer 2: xh2 = h1 @ W2 (WMMA bf16) ----
  hipMemsetAsync(agg2, 0, (size_t)N * OUTD * 4, stream);  // after GEMM1 (aliases xbf)
  {
    long long mtiles = (N + 15) >> 4;
    long long waves = mtiles * (OUTD >> 6);
    k_gemm_bf16<<<nblk(waves * 32, B), B, 0, stream>>>(h1bf, w2p, xh2, N, HID, OUTD, waves);
  }
  k_esed<<<nblk(N, B), B, 0, stream>>>(xh2, as2, ad2, es2, ed2, N, 0, 64);
  k_edge_max<<<nblk(ET, B), B, 0, stream>>>(ei, E, ET, es2, ed2, m2, 0);
  k_edge_sum<<<nblk(ET, B), B, 0, stream>>>(ei, E, ET, es2, ed2, m2, den2, 0);
  k_edge_aggr<<<nblk(ET * 32, B), B, 0, stream>>>(ei, E, ET, es2, ed2, m2, den2,
                                                  xh2, agg2, 1, 6, 2);
  k_ln_logsm<<<nblk(N * 32, B), B, 0, stream>>>(agg2, b2, g2, be2, (float*)d_out, N);
}
